// LSTM_79216376807633
// MI455X (gfx1250) — matrix operand order repaired and checked by reading, not compiled, not run on hardware
//
#include <hip/hip_runtime.h>
#include <math.h>

// ---------------- WMMA vector types (gfx1250, wave32) ----------------
typedef __attribute__((ext_vector_type(16))) __bf16 v16bf;
typedef __attribute__((ext_vector_type(8)))  float  v8f;
typedef __attribute__((ext_vector_type(4)))  int    v4i;

union Frag16 {           // 16x32 bf16 A-frag or 32x16 bf16 B-frag: 8 VGPRs
    v16bf v;
    v4i   i[2];
};

#define H 1024
#define B 64
#define T 512

// CDNA5 async copy: global -> LDS, 16B per lane, tracked by ASYNCcnt.
__device__ __forceinline__ void async_b128(const void* gsrc, const void* ldst) {
    asm volatile("global_load_async_to_lds_b128 %0, %1, off"
                 :
                 : "v"((unsigned)(size_t)ldst),
                   "v"((unsigned long long)(size_t)gsrc)
                 : "memory");
}

// fast sigmoid/tanh built on v_exp_f32 / v_rcp_f32 (saturate correctly)
__device__ __forceinline__ float fsigmoid(float x) {
    return __builtin_amdgcn_rcpf(1.0f + __expf(-x));
}
__device__ __forceinline__ float ftanh(float x) {
    return 1.0f - 2.0f * __builtin_amdgcn_rcpf(__expf(2.0f * x) + 1.0f);
}

// =====================================================================
// Prep kernels
// =====================================================================
__global__ void build_wcat_kernel(const float* __restrict__ Whh,
                                  const float* __restrict__ Wih,
                                  __bf16* __restrict__ Wcat,
                                  int kin, int kc) {
    const size_t total = (size_t)4 * H * kc;
    for (size_t i = (size_t)blockIdx.x * blockDim.x + threadIdx.x; i < total;
         i += (size_t)gridDim.x * blockDim.x) {
        const int n = (int)(i / kc);
        const int k = (int)(i % kc);
        const float v = (k < H) ? Whh[(size_t)n * H + k]
                                : Wih[(size_t)n * kin + (k - H)];
        Wcat[i] = (__bf16)v;
    }
}

__global__ void build_bias_kernel(const float* __restrict__ b0,
                                  const float* __restrict__ b1,
                                  float* __restrict__ out, int n) {
    for (int i = blockIdx.x * blockDim.x + threadIdx.x; i < n;
         i += gridDim.x * blockDim.x)
        out[i] = b0[i] + b1[i];
}

__global__ void convert_x_kernel(const float* __restrict__ src,
                                 __bf16* __restrict__ dst, size_t n) {
    for (size_t i = (size_t)blockIdx.x * blockDim.x + threadIdx.x; i < n;
         i += (size_t)gridDim.x * blockDim.x)
        dst[i] = (__bf16)src[i];
}

__global__ void zero_kernel(unsigned int* __restrict__ p, size_t nwords) {
    for (size_t i = (size_t)blockIdx.x * blockDim.x + threadIdx.x; i < nwords;
         i += (size_t)gridDim.x * blockDim.x)
        p[i] = 0u;
}

// =====================================================================
// Fused LSTM step:  g = [h_prev | x_t] @ Wcat^T + bias, gates, c/h.
//   grid: 64 blocks x 256 thr. Block owns hidden tile j0..j0+15 (x4 gates).
//   8 waves = 4 batch tiles (mt) x 2 K-halves (hh)  -> split-K x2.
//   B weights: triple-buffered async global->LDS staging, ONE barrier
//   per chunk (buffer distance 3 makes the second barrier redundant).
//   Chunk loop fully unrolled. The four gate B-fragments are pinned in
//   DISTINCT register sets via an empty asm so all 8 ds_loads issue
//   before a single dscnt wait, then the 4 WMMAs run wait-free.
// =====================================================================
template <int LAYER>
__global__ __launch_bounds__(256)
void lstm_step_kernel(const __bf16* __restrict__ Wcat,  // [4H][KC] bf16
                      const float*  __restrict__ bias,  // [4H]
                      const __bf16* __restrict__ h_in,  // [B][H] bf16
                      __bf16*       __restrict__ h_out, // [B][H] bf16
                      float*        __restrict__ c_st,  // [B][H] f32
                      const __bf16* __restrict__ xsrc,  // L0: x_bf [B][T][128]; L1: h1_all [T][B][H]
                      __bf16*       __restrict__ h1_all,// L0 out: [T][B][H]
                      float*        __restrict__ h2f,   // L1 out: [B][H] f32
                      int t) {
    constexpr int KC  = (LAYER == 0) ? (H + 128) : (2 * H);
    constexpr int KH  = KC / 2;      // K per split-K half
    constexpr int NCH = KH / 32;     // 32-K chunks per half (18 / 32)

    // [buf][half][row = gate*16 + jlocal][K: 32 used, pad to 40 (80B) banks]
    __shared__ __align__(32) __bf16 sB[3][2][64][40];
    constexpr int BUFSTRIDE = 2 * 64 * 40;   // elements per buffer

    const int tid  = threadIdx.x;
    const int lane = tid & 31;
    const int wave = tid >> 5;
    const int hl   = lane >> 4;      // half-wave
    const int l16  = lane & 15;
    const int mt   = wave & 3;       // batch tile
    const int hh   = wave >> 2;      // split-K half
    const int m0   = mt * 16;
    const int j0   = blockIdx.x * 16;
    const int j    = j0 + l16;

    // ---- hoisted per-thread staging addresses (512 x 16B ops per chunk) ----
    const __bf16* gsrc[2];
    __bf16*       ldst[2];
#pragma unroll
    for (int o = 0; o < 2; ++o) {
        const int op   = tid + o * 256;
        const int half = op >> 8;
        const int idx  = op & 255;
        const int row  = idx >> 2;          // gate*16 + jlocal
        const int seg  = idx & 3;           // 16B segment of the 64B row
        gsrc[o] = Wcat + (size_t)((row >> 4) * H + j0 + (row & 15)) * KC
                       + half * KH + seg * 8;
        ldst[o] = &sB[0][half][row][seg * 8];
    }
    auto stage = [&](int buf, int ch) {
#pragma unroll
        for (int o = 0; o < 2; ++o)
            async_b128(gsrc[o] + ch * 32, ldst[o] + buf * BUFSTRIDE);
    };

    // ---- A fragment loader (h_prev | x_t rows), direct from L2 ----
    const int arow = m0 + l16;
    const __bf16* arow_h = h_in + (size_t)arow * H;
    const __bf16* arow_x = (LAYER == 0)
        ? xsrc + ((size_t)arow * T + t) * 128
        : xsrc + ((size_t)t * B + arow) * H;

    auto loadA = [&](Frag16& a, int ch) {
        const int kg  = hh * KH + ch * 32;
        const bool inH = (kg < H);
        const __bf16* ar = inH ? (arow_h + kg) : (arow_x + (kg - H));
        a.i[0] = *reinterpret_cast<const v4i*>(ar + hl * 16);
        a.i[1] = *reinterpret_cast<const v4i*>(ar + hl * 16 + 8);
    };

    // Accumulators; bias pre-folded into the hh==0 half's C init.
    v8f acc[4];
#pragma unroll
    for (int g = 0; g < 4; ++g) {
        const float bv = (hh == 0) ? bias[g * H + j] : 0.0f;
#pragma unroll
        for (int r = 0; r < 8; ++r) acc[g][r] = bv;
    }

    auto compute = [&](int buf, const Frag16& a) {
        // load the four gate B-fragments ...
        Frag16 b0, b1, b2, b3;
        const __bf16* p0 = &sB[buf][hh][0 * 16 + l16][hl * 16];
        const __bf16* p1 = &sB[buf][hh][1 * 16 + l16][hl * 16];
        const __bf16* p2 = &sB[buf][hh][2 * 16 + l16][hl * 16];
        const __bf16* p3 = &sB[buf][hh][3 * 16 + l16][hl * 16];
        b0.i[0] = reinterpret_cast<const v4i*>(p0)[0];
        b0.i[1] = reinterpret_cast<const v4i*>(p0)[1];
        b1.i[0] = reinterpret_cast<const v4i*>(p1)[0];
        b1.i[1] = reinterpret_cast<const v4i*>(p1)[1];
        b2.i[0] = reinterpret_cast<const v4i*>(p2)[0];
        b2.i[1] = reinterpret_cast<const v4i*>(p2)[1];
        b3.i[0] = reinterpret_cast<const v4i*>(p3)[0];
        b3.i[1] = reinterpret_cast<const v4i*>(p3)[1];
        // ... pin all four in distinct registers: forces the 8 ds_loads to
        // issue before ONE dscnt wait; the 4 WMMAs then run wait-free.
        asm volatile("" : "+v"(b0.v), "+v"(b1.v), "+v"(b2.v), "+v"(b3.v));
        acc[0] = __builtin_amdgcn_wmma_f32_16x16x32_bf16(
            false, a.v, false, b0.v, (short)0, acc[0], false, false);
        acc[1] = __builtin_amdgcn_wmma_f32_16x16x32_bf16(
            false, a.v, false, b1.v, (short)0, acc[1], false, false);
        acc[2] = __builtin_amdgcn_wmma_f32_16x16x32_bf16(
            false, a.v, false, b2.v, (short)0, acc[2], false, false);
        acc[3] = __builtin_amdgcn_wmma_f32_16x16x32_bf16(
            false, a.v, false, b3.v, (short)0, acc[3], false, false);
    };

    // ---- software pipeline: 3 LDS buffers, 1 barrier/chunk, A 2 ahead ----
    Frag16 a[3];
    stage(0, 0);
    stage(1, 1);
    loadA(a[0], 0);
    loadA(a[1], 1);

#pragma unroll
    for (int ch = 0; ch < NCH; ++ch) {
        // ensure our stage(ch) completed (in-order; stage(ch+1) may fly)
        if (ch + 1 < NCH)
            asm volatile("s_wait_asynccnt 0x2" ::: "memory");
        else
            asm volatile("s_wait_asynccnt 0x0" ::: "memory");
        asm volatile("s_barrier_signal -1" ::: "memory");
        if (ch + 2 < NCH) loadA(a[(ch + 2) % 3], ch + 2); // overlap barrier
        asm volatile("s_barrier_wait -1" ::: "memory");
        // buffer distance 3: readers of (ch+2)%3 retired at this barrier
        if (ch + 2 < NCH) stage((ch + 2) % 3, ch + 2);
        compute(ch % 3, a[ch % 3]);
    }
    __syncthreads();   // all computes done; sB reusable as reduction scratch

    // ---- split-K reduction through LDS (16KB of the 30KB sB) ----
    float* red = reinterpret_cast<float*>(&sB[0][0][0][0]);
    if (hh == 1) {
#pragma unroll
        for (int g = 0; g < 4; ++g)
            *reinterpret_cast<v8f*>(red + ((size_t)((mt * 4 + g) * 32 + lane) * 8)) = acc[g];
    }
    __syncthreads();
    if (hh == 0) {
#pragma unroll
        for (int g = 0; g < 4; ++g)
            acc[g] += *reinterpret_cast<v8f*>(red + ((size_t)((mt * 4 + g) * 32 + lane) * 8));

        // ---- fused gates: i/f/g/o elementwise-aligned across acc[0..3] ----
#pragma unroll
        for (int r = 0; r < 8; ++r) {
            const int m = m0 + r + 8 * hl;
            const size_t idx = (size_t)m * H + j;
            const float ig = fsigmoid(acc[0][r]);
            const float fg = fsigmoid(acc[1][r]);
            const float gc = ftanh(acc[2][r]);
            const float og = fsigmoid(acc[3][r]);
            const float cn = fg * c_st[idx] + ig * gc;
            c_st[idx] = cn;
            const float hn = og * ftanh(cn);
            h_out[idx] = (__bf16)hn;
            if (LAYER == 0) {
                h1_all[((size_t)t * B + m) * H + j] = (__bf16)hn;
            } else {
                h2f[idx] = hn;
            }
        }
    }
}

// =====================================================================
// Final FC: out[b] = dot(h2[b], fc_w) + fc_b   (64 outputs)
// =====================================================================
__global__ __launch_bounds__(256)
void fc_kernel(const float* __restrict__ h2, const float* __restrict__ w,
               const float* __restrict__ b, float* __restrict__ out) {
    __shared__ float red[256];
    const int bi = blockIdx.x;
    float s = 0.f;
    for (int k = threadIdx.x; k < H; k += 256) s += h2[(size_t)bi * H + k] * w[k];
    red[threadIdx.x] = s;
    __syncthreads();
    for (int off = 128; off > 0; off >>= 1) {
        if ((int)threadIdx.x < off) red[threadIdx.x] += red[threadIdx.x + off];
        __syncthreads();
    }
    if (threadIdx.x == 0) out[bi] = red[0] + b[0];
}

// =====================================================================
extern "C" void kernel_launch(void* const* d_in, const int* in_sizes, int n_in,
                              void* d_out, int out_size, void* d_ws, size_t ws_size,
                              hipStream_t stream) {
    const float* x     = (const float*)d_in[0];
    const float* W_ih0 = (const float*)d_in[1];
    const float* W_hh0 = (const float*)d_in[2];
    const float* b_ih0 = (const float*)d_in[3];
    const float* b_hh0 = (const float*)d_in[4];
    const float* W_ih1 = (const float*)d_in[5];
    const float* W_hh1 = (const float*)d_in[6];
    const float* b_ih1 = (const float*)d_in[7];
    const float* b_hh1 = (const float*)d_in[8];
    const float* fc_w  = (const float*)d_in[9];
    const float* fc_b  = (const float*)d_in[10];
    float* out = (float*)d_out;

    // ---- workspace carve-up (all offsets 256B aligned) ----
    char* ws = (char*)d_ws;
    size_t off = 0;
    auto take = [&](size_t bytes) { char* p = ws + off; off += (bytes + 255) & ~(size_t)255; return p; };

    __bf16* W0cat  = (__bf16*)take((size_t)4 * H * (H + 128) * 2); //  9.4 MB
    __bf16* W1cat  = (__bf16*)take((size_t)4 * H * (2 * H) * 2);   // 16.8 MB
    float*  bias0  = (float*) take((size_t)4 * H * 4);
    float*  bias1  = (float*) take((size_t)4 * H * 4);
    __bf16* x_bf   = (__bf16*)take((size_t)B * T * 128 * 2);       //  8.4 MB
    __bf16* h1_all = (__bf16*)take((size_t)T * B * H * 2);         // 64   MB
    // contiguous zero region: hA0, hP0, c0, hA1, hP1, c1
    char*   zbase  = ws + off;
    __bf16* hA0 = (__bf16*)take((size_t)B * H * 2);
    __bf16* hP0 = (__bf16*)take((size_t)B * H * 2);
    float*  c0  = (float*) take((size_t)B * H * 4);
    __bf16* hA1 = (__bf16*)take((size_t)B * H * 2);
    __bf16* hP1 = (__bf16*)take((size_t)B * H * 2);
    float*  c1  = (float*) take((size_t)B * H * 4);
    size_t zwords = ((size_t)(ws + off - zbase)) / 4;
    float*  h2f = (float*) take((size_t)B * H * 4);
    (void)ws_size; (void)in_sizes; (void)n_in; (void)out_size;

    // ---- prep (runs every call; ws is not preserved across calls) ----
    build_wcat_kernel<<<1024, 256, 0, stream>>>(W_hh0, W_ih0, W0cat, 128, H + 128);
    build_wcat_kernel<<<1024, 256, 0, stream>>>(W_hh1, W_ih1, W1cat, H, 2 * H);
    build_bias_kernel<<<16, 256, 0, stream>>>(b_ih0, b_hh0, bias0, 4 * H);
    build_bias_kernel<<<16, 256, 0, stream>>>(b_ih1, b_hh1, bias1, 4 * H);
    convert_x_kernel<<<2048, 256, 0, stream>>>(x, x_bf, (size_t)B * T * 128);
    zero_kernel<<<256, 256, 0, stream>>>((unsigned int*)zbase, zwords);

    // ---- layer 0: 512 serial steps (h ping-pong across launches) ----
    for (int t = 0; t < T; ++t) {
        const __bf16* hin  = (t & 1) ? hP0 : hA0;
        __bf16*       hout = (t & 1) ? hA0 : hP0;
        lstm_step_kernel<0><<<64, 256, 0, stream>>>(W0cat, bias0, hin, hout, c0,
                                                    x_bf, h1_all, nullptr, t);
    }
    // ---- layer 1: 512 serial steps consuming h1_all ----
    for (int t = 0; t < T; ++t) {
        const __bf16* hin  = (t & 1) ? hP1 : hA1;
        __bf16*       hout = (t & 1) ? hA1 : hP1;
        lstm_step_kernel<1><<<64, 256, 0, stream>>>(W1cat, bias1, hin, hout, c1,
                                                    h1_all, nullptr, h2f, t);
    }

    // ---- final FC on h2[:, T-1, :] ----
    fc_kernel<<<B, 256, 0, stream>>>(h2f, fc_w, fc_b, out);
}
